// EvalNet_81260781240520
// MI455X (gfx1250) — matrix operand order repaired and checked by reading, not compiled, not run on hardware
//
#include <hip/hip_runtime.h>

typedef __attribute__((ext_vector_type(4)))  float    v4f;
typedef __attribute__((ext_vector_type(8)))  float    v8f;
typedef __attribute__((ext_vector_type(4)))  _Float16 v4h;
typedef __attribute__((ext_vector_type(8)))  _Float16 v8h;
typedef __attribute__((ext_vector_type(16))) _Float16 v16h;

#define D          1024     // HIDDEN
#define ROWS       769      // emb table rows (INPUT_DIM + 1)
#define SLOTS      32       // indices per sample
#define H2         32       // HIDDEN2
#define NB         8        // buckets
#define WPB        4        // waves per block (128 threads)
#define ACH_STRIDE 72       // padded f16 stride of A-chunk row (144B: bank-spread)
#define H2_STRIDE  33       // padded f32 stride of h2 row
#define FC2_STRIDE 1032     // padded f16 stride of fc2 row (2064B: B-frag reads conflict-free)

// ---- prep: one-time (per launch) f32 -> f16 conversion of emb_w and fc2_w ----
__global__ void cvt_prep(const float* __restrict__ emb, const float* __restrict__ fc2,
                         _Float16* __restrict__ embh, _Float16* __restrict__ fc2h,
                         int nemb, int nfc2)
{
    const int i      = blockIdx.x * blockDim.x + threadIdx.x;
    const int stride = gridDim.x * blockDim.x;
    for (int j = i; j < nemb; j += stride) embh[j] = (_Float16)emb[j];
    for (int j = i; j < nfc2; j += stride) fc2h[j] = (_Float16)fc2[j];
}

__global__ __launch_bounds__(WPB * 32)
void evalnet_fused(const int* __restrict__ x, const int* __restrict__ piece_count,
                   const _Float16* __restrict__ emb_h, const float* __restrict__ bias1,
                   const _Float16* __restrict__ fc2_h, const float* __restrict__ fc2_b,
                   const float* __restrict__ cp_w,  const float* __restrict__ cp_b,
                   const float* __restrict__ wdl_w, const float* __restrict__ wdl_b,
                   float* __restrict__ out, int B)
{
    __shared__ int s_idx[WPB][16 * SLOTS];                          // 8 KB
    __shared__ __align__(16) _Float16 s_ach[WPB][16 * ACH_STRIDE];  // 9 KB
    __shared__ float s_h2[WPB][16 * H2_STRIDE];                     // 8.25 KB
    __shared__ __align__(16) _Float16 s_fc2[H2 * FC2_STRIDE];       // 64.5 KB, shared by block

    const int lane = threadIdx.x & 31;
    const int wave = threadIdx.x >> 5;
    const int tile = blockIdx.x * WPB + wave;   // one 16-row batch tile per wave
    const int row0 = tile * 16;

    int*      idx_w = s_idx[wave];
    _Float16* ach_w = s_ach[wave];
    float*    h2_w  = s_h2[wave];

    // ---- stage fc2 (f16) into LDS: 4 threads per row, 256 halves each ----
    {
        const int r  = threadIdx.x >> 2;          // 0..31
        const int c0 = (threadIdx.x & 3) * 256;   // 0,256,512,768
        const _Float16* src = fc2_h + r * D + c0;
        _Float16*       dst = s_fc2 + r * FC2_STRIDE + c0;
        #pragma unroll
        for (int j = 0; j < 32; ++j)
            *(v8h*)(dst + j * 8) = *(const v8h*)(src + j * 8);
    }

    // ---- stage this tile's 512 indices into LDS (contiguous in x) ----
    const int* xs = x + row0 * SLOTS;
    #pragma unroll
    for (int i = 0; i < 16; ++i) idx_w[lane + 32 * i] = xs[lane + 32 * i];
    __syncthreads();

    const int col = lane & 15;   // float4 column within chunk / N or M sub-index
    const int lhi = lane >> 4;   // lane half (WMMA fragment phase)

    v8f c0 = {}; v8f c1 = {};    // C fragments: [16 x 32] accumulator

    #pragma unroll 1
    for (int kc = 0; kc < 16; ++kc) {            // K chunks of 64
        const int kbase = kc * 64;

        // ---- f16 gather-sum (v_pk_add_f16) + bias + screlu -> A-chunk in LDS ----
        const _Float16* ecol = emb_h + kbase + col * 4;
        const v4f bf = *(const v4f*)(bias1 + kbase + col * 4);
        v4h bias_h;
        #pragma unroll
        for (int j = 0; j < 4; ++j) bias_h[j] = (_Float16)bf[j];

        #pragma unroll 2
        for (int i = 0; i < 8; ++i) {
            const int row = lhi + 2 * i;
            const int* ip = idx_w + row * SLOTS;
            v4h acc = {};
            #pragma unroll
            for (int k = 0; k < SLOTS; ++k)
                acc += *(const v4h*)(ecol + ip[k] * D);   // pad row is all-zero
            acc += bias_h;
            #pragma unroll
            for (int j = 0; j < 4; ++j) {
                _Float16 v = acc[j];
                v = v < (_Float16)0.f ? (_Float16)0.f : (v > (_Float16)1.f ? (_Float16)1.f : v);
                acc[j] = v * v;
            }
            *(v4h*)(ach_w + row * ACH_STRIDE + col * 4) = acc;
        }
        __syncthreads();

        // ---- 2 K-steps x 2 N-tiles of v_wmma_f32_16x16x32_f16 ----
        #pragma unroll
        for (int s = 0; s < 2; ++s) {
            // A frag (16x32 f16 layout): lane<16 holds K{0..7,16..23}, lane>=16 +8
            const _Float16* pa = ach_w + col * ACH_STRIDE + s * 32 + lhi * 8;
            v8h alo = *(const v8h*)(pa);        // K +0..7  (ds_load_b128)
            v8h ahi = *(const v8h*)(pa + 16);   // K +16..23
            v16h a = __builtin_shufflevector(alo, ahi,
                     0,1,2,3,4,5,6,7,8,9,10,11,12,13,14,15);
            #pragma unroll
            for (int t = 0; t < 2; ++t) {
                // B frag (32x16 f16): lane n holds col N=n, K{0..7,16..23}, lane>=16 +8
                const int n = t * 16 + col;
                const _Float16* bp = s_fc2 + n * FC2_STRIDE + kbase + s * 32 + lhi * 8;
                v8h blo = *(const v8h*)(bp);
                v8h bhi = *(const v8h*)(bp + 16);
                v16h b = __builtin_shufflevector(blo, bhi,
                         0,1,2,3,4,5,6,7,8,9,10,11,12,13,14,15);
                if (t == 0)
                    c0 = __builtin_amdgcn_wmma_f32_16x16x32_f16(
                             false, a, false, b, (short)0, c0, false, false);
                else
                    c1 = __builtin_amdgcn_wmma_f32_16x16x32_f16(
                             false, a, false, b, (short)0, c1, false, false);
            }
        }
        __syncthreads();   // A-chunk reads done before next chunk overwrites
    }

    // ---- h2 = screlu(C + fc2_b) -> LDS [16 x 32] ----
    #pragma unroll
    for (int t = 0; t < 2; ++t) {
        const int n = t * 16 + col;
        const float bb = fc2_b[n];
        const v8f cc = (t == 0) ? c0 : c1;
        #pragma unroll
        for (int r = 0; r < 8; ++r) {
            const int m = r + lhi * 8;          // C/D layout: VGPR r -> M=r (+8 for hi lanes)
            float v = cc[r] + bb;
            v = v < 0.f ? 0.f : (v > 1.f ? 1.f : v);
            h2_w[m * H2_STRIDE + n] = v * v;
        }
    }
    __syncthreads();

    // ---- bucketed heads: 64 outputs per tile, 2 tasks per lane ----
    #pragma unroll
    for (int tsk = 0; tsk < 2; ++tsk) {
        const int task = lane + tsk * 32;
        const int row  = task >> 2;             // 0..15
        const int o    = task & 3;              // 0=cp, 1..3=wdl
        const int gr   = row0 + row;
        const int pc   = piece_count[gr];
        int bkt = (pc - 2) * NB / 30;
        bkt = bkt < 0 ? 0 : (bkt > NB - 1 ? NB - 1 : bkt);
        const float* w; float b;
        if (o == 0) { w = cp_w + bkt * H2; b = cp_b[bkt]; }
        else { const int r = bkt * 3 + (o - 1); w = wdl_w + r * H2; b = wdl_b[r]; }
        float s = b;
        const float* hr = h2_w + row * H2_STRIDE;
        #pragma unroll
        for (int k = 0; k < H2; ++k) s += hr[k] * w[k];
        if (o == 0) out[gr] = s;
        else        out[B + gr * 3 + (o - 1)] = s;
    }
}

extern "C" void kernel_launch(void* const* d_in, const int* in_sizes, int n_in,
                              void* d_out, int out_size, void* d_ws, size_t ws_size,
                              hipStream_t stream) {
    const int*   x   = (const int*)d_in[0];
    const int*   pc  = (const int*)d_in[1];
    const float* emb = (const float*)d_in[2];
    const float* b1  = (const float*)d_in[3];
    const float* f2w = (const float*)d_in[4];
    const float* f2b = (const float*)d_in[5];
    const float* cpw = (const float*)d_in[6];
    const float* cpb = (const float*)d_in[7];
    const float* wdw = (const float*)d_in[8];
    const float* wdb = (const float*)d_in[9];
    float* out = (float*)d_out;

    const int B    = in_sizes[1];        // 16384
    const int nemb = ROWS * D;           // 787,456
    const int nfc2 = H2 * D;             // 32,768

    // d_ws layout: [embh: nemb f16][fc2h: nfc2 f16]  (~1.64 MB)
    _Float16* embh = (_Float16*)d_ws;
    _Float16* fc2h = embh + nemb;

    cvt_prep<<<512, 256, 0, stream>>>(emb, f2w, embh, fc2h, nemb, nfc2);

    const int tiles  = B / 16;           // 1024
    const int blocks = tiles / WPB;      // 256 blocks x 128 threads
    evalnet_fused<<<blocks, WPB * 32, 0, stream>>>(
        x, pc, embh, b1, fc2h, f2b, cpw, cpb, wdw, wdb, out, B);
}
